// Actor_74423193305350
// MI455X (gfx1250) — compile-verified, matrix operand order checked by reading, not run on hardware
//
#include <hip/hip_runtime.h>
#include <hip/hip_bf16.h>

#define N_NODES 50000
#define N_EDGES 400000
#define N_GRAPHS 16
#define HID 64
#define N_LAYERS 3
#define BN_EPS 1e-5f

typedef __attribute__((ext_vector_type(16))) _Float16 v16h;
typedef __attribute__((ext_vector_type(8)))  _Float16 v8h;
typedef __attribute__((ext_vector_type(8)))  float    v8f;

// ---------------- WMMA wave-tile GEMM (f16 operands, f32 accum) ----------------------------
// Fragment layout (16-bit A 16x32 / B 32x16): lane L: M(or N) = L&15 ; kb = k0 + (L>>4)*8
//   f[0..7]  = K kb..kb+7   (contiguous)      f[8..15] = K kb+16..kb+23 (contiguous)
// => each fragment is two 128-bit loads from a 64-half row.
__device__ __forceinline__ v16h load_frag_f16(const _Float16* __restrict__ rowbase,
                                              int k0, int lane) {
    int kb = k0 + ((lane >> 4) << 3);
    v8h lo = *(const v8h*)(rowbase + kb);
    v8h hi = *(const v8h*)(rowbase + kb + 16);
    v16h f;
#pragma unroll
    for (int i = 0; i < 8; ++i) { f[i] = lo[i]; f[8 + i] = hi[i]; }
    return f;
}

// One wave computes D[16x16] = X[m0:m0+16, 0:64] * W[64, n0:n0+16] + bias
// x : row-major f16 [rows][64]      wt : f16 transposed weights [n][k] (64x64)
__device__ __forceinline__ void wave_gemm_tile(const _Float16* __restrict__ x, int m0,
                                               const _Float16* __restrict__ wt,
                                               const float* __restrict__ bias,
                                               int n0, float* __restrict__ out) {
    int lane = threadIdx.x & 31;
    const _Float16* arow = x  + (size_t)(m0 + (lane & 15)) * HID;
    const _Float16* brow = wt + (size_t)(n0 + (lane & 15)) * HID;
    v8f acc = {};
    acc = __builtin_amdgcn_wmma_f32_16x16x32_f16(false, load_frag_f16(arow, 0, lane),
                                                 false, load_frag_f16(brow, 0, lane),
                                                 (short)0, acc, false, false);
    acc = __builtin_amdgcn_wmma_f32_16x16x32_f16(false, load_frag_f16(arow, 32, lane),
                                                 false, load_frag_f16(brow, 32, lane),
                                                 (short)0, acc, false, false);
    float bv = bias[n0 + (lane & 15)];
    // C/D layout: VGPR r, lanes 0-15 -> M=r, lanes 16-31 -> M=r+8 ; N = lane&15
#pragma unroll
    for (int r = 0; r < 8; ++r) {
        int row = m0 + r + ((lane >> 4) << 3);
        out[(size_t)row * HID + n0 + (lane & 15)] = acc[r] + bv;
    }
}

// 16 waves/block: wave -> (matrix in {A,B,D,E}, n-tile).  wt4 = 4 transposed f16 weights.
__global__ __launch_bounds__(512) void gemm4_node_kernel(
    const _Float16* __restrict__ hh,
    const _Float16* __restrict__ wtA, const float* __restrict__ bA, float* __restrict__ Ah,
    const _Float16* __restrict__ wtB, const float* __restrict__ bB, float* __restrict__ Bh,
    const _Float16* __restrict__ wtD, const float* __restrict__ bD, float* __restrict__ Dh,
    const _Float16* __restrict__ wtE, const float* __restrict__ bE, float* __restrict__ Eh) {
    int wave = threadIdx.x >> 5;
    int mat = wave >> 2, nt = wave & 3;
    int m0 = blockIdx.x << 4;
    const _Float16* w; const float* bb; float* out;
    if (mat == 0)      { w = wtA; bb = bA; out = Ah; }
    else if (mat == 1) { w = wtB; bb = bB; out = Bh; }
    else if (mat == 2) { w = wtD; bb = bD; out = Dh; }
    else               { w = wtE; bb = bE; out = Eh; }
    wave_gemm_tile(hh, m0, w, bb, nt << 4, out);
}

// 4 waves/block, one n-tile each
__global__ __launch_bounds__(128) void gemm_edge_kernel(
    const _Float16* __restrict__ ee, const _Float16* __restrict__ wtC,
    const float* __restrict__ bC, float* __restrict__ Ce) {
    int wave = threadIdx.x >> 5;
    wave_gemm_tile(ee, blockIdx.x << 4, wtC, bC, wave << 4, Ce);
}

// transpose+convert all 15 weight matrices to f16 [mat][layer][n][k]
__global__ void cvt_weights_kernel(const float* __restrict__ A, const float* __restrict__ B,
                                   const float* __restrict__ C, const float* __restrict__ D,
                                   const float* __restrict__ E, _Float16* __restrict__ wt) {
    int i = blockIdx.x * 256 + threadIdx.x;
    if (i >= 5 * N_LAYERS * HID * HID) return;
    int mat = i / (N_LAYERS * HID * HID);
    int rem = i % (N_LAYERS * HID * HID);
    int l  = rem / (HID * HID);
    int nk = rem % (HID * HID);
    int n = nk >> 6, k = nk & 63;
    const float* w = (mat == 0) ? A : (mat == 1) ? B : (mat == 2) ? C : (mat == 3) ? D : E;
    wt[i] = (_Float16)w[l * HID * HID + k * HID + n];
}

// ---------------- elementwise / scatter kernels --------------------------------------------
__global__ void embed_kernel(const float* __restrict__ in, int in_dim,
                             const float* __restrict__ w, const float* __restrict__ b,
                             float* __restrict__ out, _Float16* __restrict__ out16,
                             int nrows, int recip) {
    int i = blockIdx.x * 256 + threadIdx.x;
    if (i >= nrows * HID) return;
    int r = i >> 6, f = i & 63;
    float s = b[f];
    for (int k = 0; k < in_dim; ++k) {
        float v = in[(size_t)r * in_dim + k];
        if (recip) v = 1.0f / v;
        s += v * w[k * HID + f];
    }
    out[i] = s;
    out16[i] = (_Float16)s;
}

// e_hat = Dh[src] + Eh[dst] + Ce (in-place into Ce); sigma scatter-adds to num/den
__global__ void edge_kernel(const float* __restrict__ Dh, const float* __restrict__ Eh,
                            const float* __restrict__ Bh, float* __restrict__ Ce,
                            const int* __restrict__ src, const int* __restrict__ dst,
                            float* __restrict__ num, float* __restrict__ den) {
    int i = blockIdx.x * 256 + threadIdx.x;
    if (i >= N_EDGES * HID) return;
    int edge = i >> 6, f = i & 63;
    int s = src[edge], d = dst[edge];
    float eh = Dh[(size_t)s * HID + f] + Eh[(size_t)d * HID + f] + Ce[i];
    Ce[i] = eh;
    float sig = 1.0f / (1.0f + __expf(-eh));
    atomicAdd(&num[(size_t)d * HID + f], sig * Bh[(size_t)s * HID + f]);
    atomicAdd(&den[(size_t)d * HID + f], sig);
}

// h_new = Ah + num/(den+1e-6), in-place into Ah
__global__ void hnew_kernel(float* __restrict__ Ah, const float* __restrict__ num,
                            const float* __restrict__ den) {
    int i = blockIdx.x * 256 + threadIdx.x;
    if (i >= N_NODES * HID) return;
    Ah[i] += num[i] / (den[i] + 1e-6f);
}

// per-feature sum / sumsq over rows (grid-stride; stride multiple of 64 keeps f fixed)
__global__ __launch_bounds__(256) void stats_kernel(const float* __restrict__ x, int nrows,
                                                    float* __restrict__ stats) {
    __shared__ float ssum[HID], ssq[HID];
    int t = threadIdx.x, f = t & 63;
    if (t < HID) { ssum[t] = 0.f; ssq[t] = 0.f; }
    __syncthreads();
    float s = 0.f, q = 0.f;
    int total = nrows * HID;
    for (int i = blockIdx.x * 256 + t; i < total; i += gridDim.x * 256) {
        float v = x[i];
        s += v; q += v * v;
    }
    atomicAdd(&ssum[f], s);
    atomicAdd(&ssq[f], q);
    __syncthreads();
    if (t < HID) {
        atomicAdd(&stats[t], ssum[t]);
        atomicAdd(&stats[HID + t], ssq[t]);
    }
}

// base += relu(bn(xnew)); also refresh the f16 shadow used by the next layer's WMMA GEMM
__global__ void bn_apply_kernel(float* __restrict__ base, _Float16* __restrict__ base16,
                                const float* __restrict__ xnew,
                                const float* __restrict__ stats, const float* __restrict__ g,
                                const float* __restrict__ b, int nrows) {
    int i = blockIdx.x * 256 + threadIdx.x;
    if (i >= nrows * HID) return;
    int f = i & 63;
    float inv = 1.0f / (float)nrows;
    float mu  = stats[f] * inv;
    float var = stats[HID + f] * inv - mu * mu;
    float v = (xnew[i] - mu) * rsqrtf(var + BN_EPS) * g[f] + b[f];
    float r = base[i] + fmaxf(v, 0.0f);
    base[i] = r;
    base16[i] = (_Float16)r;
}

__global__ void readout_kernel(const float* __restrict__ h, const int* __restrict__ gid,
                               float* __restrict__ hg, float* __restrict__ cnt) {
    int i = blockIdx.x * 256 + threadIdx.x;
    if (i >= N_NODES * HID) return;
    int node = i >> 6, f = i & 63;
    int g = gid[node];
    atomicAdd(&hg[g * HID + f], h[i]);
    if (f == 0) atomicAdd(&cnt[g], 1.0f);
}

__global__ __launch_bounds__(256) void mlp_kernel(
    const float* __restrict__ hg, const float* __restrict__ cnt, const float* __restrict__ state,
    const float* __restrict__ w1, const float* __restrict__ b1,
    const float* __restrict__ w2, const float* __restrict__ b2,
    const float* __restrict__ w3, const float* __restrict__ b3,
    float* __restrict__ out) {
    __shared__ float x[N_GRAPHS][68];
    __shared__ float y1[N_GRAPHS][256];
    __shared__ float y2[N_GRAPHS][256];
    int t = threadIdx.x;
    for (int i = t; i < N_GRAPHS * 68; i += 256) {
        int g = i / 68, c = i % 68;
        x[g][c] = (c < HID) ? hg[g * HID + c] / cnt[g] : state[g * 4 + (c - HID)];
    }
    __syncthreads();
    for (int i = t; i < N_GRAPHS * 256; i += 256) {
        int g = i >> 8, o = i & 255;
        float s = b1[o];
        for (int k = 0; k < 68; ++k) s += x[g][k] * w1[k * 256 + o];
        y1[g][o] = fmaxf(s, 0.0f);
    }
    __syncthreads();
    for (int i = t; i < N_GRAPHS * 256; i += 256) {
        int g = i >> 8, o = i & 255;
        float s = b2[o];
        for (int k = 0; k < 256; ++k) s += y1[g][k] * w2[k * 256 + o];
        y2[g][o] = fmaxf(s, 0.0f);
    }
    __syncthreads();
    if (t < N_GRAPHS * 2) {
        int g = t >> 1, o = t & 1;
        float s = b3[o];
        for (int k = 0; k < 256; ++k) s += y2[g][k] * w3[k * 2 + o];
        out[g * 2 + o] = tanhf(s);  // MAX_ACTION = 1.0
    }
}

// ---------------- host launcher -------------------------------------------------------------
extern "C" void kernel_launch(void* const* d_in, const int* in_sizes, int n_in,
                              void* d_out, int out_size, void* d_ws, size_t ws_size,
                              hipStream_t stream) {
    const float* state   = (const float*)d_in[0];
    const float* h_feat  = (const float*)d_in[1];
    const float* e_feat  = (const float*)d_in[2];
    const int*   src     = (const int*)d_in[3];
    const int*   dst     = (const int*)d_in[4];
    const int*   gid     = (const int*)d_in[5];
    const float* emb_h_w = (const float*)d_in[6];
    const float* emb_h_b = (const float*)d_in[7];
    const float* emb_e_w = (const float*)d_in[8];
    const float* emb_e_b = (const float*)d_in[9];
    const float* A_w = (const float*)d_in[10]; const float* A_b = (const float*)d_in[11];
    const float* B_w = (const float*)d_in[12]; const float* B_b = (const float*)d_in[13];
    const float* C_w = (const float*)d_in[14]; const float* C_b = (const float*)d_in[15];
    const float* D_w = (const float*)d_in[16]; const float* D_b = (const float*)d_in[17];
    const float* E_w = (const float*)d_in[18]; const float* E_b = (const float*)d_in[19];
    const float* bn_h_g = (const float*)d_in[20]; const float* bn_h_b = (const float*)d_in[21];
    const float* bn_e_g = (const float*)d_in[22]; const float* bn_e_b = (const float*)d_in[23];
    const float* l1_w = (const float*)d_in[24]; const float* l1_b = (const float*)d_in[25];
    const float* l2_w = (const float*)d_in[26]; const float* l2_b = (const float*)d_in[27];
    const float* l3_w = (const float*)d_in[28]; const float* l3_b = (const float*)d_in[29];
    float* out = (float*)d_out;

    // workspace layout (all chunks 256B aligned)
    const size_t NH = (size_t)N_NODES * HID;   // 3.2M
    const size_t EH = (size_t)N_EDGES * HID;   // 25.6M
    const size_t WT = 5 * N_LAYERS * HID * HID;
    float* ws = (float*)d_ws;
    float* h     = ws;            ws += NH;
    float* Ah    = ws;            ws += NH;
    float* Bh    = ws;            ws += NH;
    float* Dh    = ws;            ws += NH;
    float* Eh    = ws;            ws += NH;
    float* num   = ws;            ws += NH;
    float* den   = ws;            ws += NH;
    float* e     = ws;            ws += EH;
    float* Ce    = ws;            ws += EH;   // reused as e_hat
    float* stats = ws;            ws += 256;  // [0:128) h-stats, [128:256) e-stats
    float* hg    = ws;            ws += N_GRAPHS * HID;
    float* cnt   = ws;            ws += 64;
    _Float16* hh   = (_Float16*)ws;  ws += NH / 2;   // f16 shadow of h
    _Float16* ee   = (_Float16*)ws;  ws += EH / 2;   // f16 shadow of e
    _Float16* wt16 = (_Float16*)ws;  ws += WT / 2;   // transposed f16 weights [mat][l][n][k]

    const int nNH = (int)NH, nEH = (int)EH;
    dim3 blk(256);

    cvt_weights_kernel<<<((int)WT + 255) / 256, blk, 0, stream>>>(A_w, B_w, C_w, D_w, E_w, wt16);

    embed_kernel<<<(nNH + 255) / 256, blk, 0, stream>>>(h_feat, 6, emb_h_w, emb_h_b, h, hh, N_NODES, 0);
    embed_kernel<<<(nEH + 255) / 256, blk, 0, stream>>>(e_feat, 2, emb_e_w, emb_e_b, e, ee, N_EDGES, 1);

    for (int l = 0; l < N_LAYERS; ++l) {
        const int WSZ = HID * HID;
        const _Float16* wtA = wt16 + (0 * N_LAYERS + l) * WSZ; const float* bA = A_b + l * HID;
        const _Float16* wtB = wt16 + (1 * N_LAYERS + l) * WSZ; const float* bB = B_b + l * HID;
        const _Float16* wtC = wt16 + (2 * N_LAYERS + l) * WSZ; const float* bC = C_b + l * HID;
        const _Float16* wtD = wt16 + (3 * N_LAYERS + l) * WSZ; const float* bD = D_b + l * HID;
        const _Float16* wtE = wt16 + (4 * N_LAYERS + l) * WSZ; const float* bE = E_b + l * HID;

        gemm4_node_kernel<<<N_NODES / 16, 512, 0, stream>>>(hh, wtA, bA, Ah, wtB, bB, Bh,
                                                            wtD, bD, Dh, wtE, bE, Eh);
        gemm_edge_kernel<<<N_EDGES / 16, 128, 0, stream>>>(ee, wtC, bC, Ce);

        hipMemsetAsync(num, 0, NH * sizeof(float), stream);
        hipMemsetAsync(den, 0, NH * sizeof(float), stream);
        edge_kernel<<<(nEH + 255) / 256, blk, 0, stream>>>(Dh, Eh, Bh, Ce, src, dst, num, den);
        hnew_kernel<<<(nNH + 255) / 256, blk, 0, stream>>>(Ah, num, den);

        hipMemsetAsync(stats, 0, 256 * sizeof(float), stream);
        stats_kernel<<<512, 256, 0, stream>>>(Ah, N_NODES, stats);
        stats_kernel<<<512, 256, 0, stream>>>(Ce, N_EDGES, stats + 128);
        bn_apply_kernel<<<(nNH + 255) / 256, blk, 0, stream>>>(h, hh, Ah, stats,
                                                               bn_h_g + l * HID, bn_h_b + l * HID, N_NODES);
        bn_apply_kernel<<<(nEH + 255) / 256, blk, 0, stream>>>(e, ee, Ce, stats + 128,
                                                               bn_e_g + l * HID, bn_e_b + l * HID, N_EDGES);
    }

    hipMemsetAsync(hg, 0, (N_GRAPHS * HID + 64) * sizeof(float), stream);
    readout_kernel<<<(nNH + 255) / 256, blk, 0, stream>>>(h, gid, hg, cnt);
    mlp_kernel<<<1, 256, 0, stream>>>(hg, cnt, state, l1_w, l1_b, l2_w, l2_b, l3_w, l3_b, out);
}
